// C2FPromptPruner_Embedding_33088428049155
// MI455X (gfx1250) — matrix operand order repaired and checked by reading, not compiled, not run on hardware
//
#include <hip/hip_runtime.h>
#include <hip/hip_bf16.h>

#define D 4096
#define NSHOTS 32
#define SEQ 512
#define TGT_SHOT 4
#define TOKROWS (TGT_SHOT * SEQ)       // 2048
#define TGT_TOK (TGT_SHOT * 256)       // 1024

typedef __attribute__((ext_vector_type(16))) __bf16 v16bf;
typedef __attribute__((ext_vector_type(8)))  __bf16 v8bf;
typedef __attribute__((ext_vector_type(8)))  float  v8f;

static __device__ __forceinline__ unsigned short f32_to_bf16(float f) {
    unsigned int u = __float_as_uint(f);
    unsigned int r = (u + 0x7FFFu + ((u >> 16) & 1u)) >> 16;   // RNE
    return (unsigned short)r;
}

// Low 32 bits of a generic pointer to LDS == LDS byte offset (aperture scheme).
static __device__ __forceinline__ unsigned lds_off(const void* p) {
    return (unsigned)(unsigned long long)p;
}

// async copy 16B global -> LDS, tracked by ASYNCcnt (gfx1250)
static __device__ __forceinline__ void async_b128(unsigned lds, const void* g) {
    asm volatile("global_load_async_to_lds_b128 %0, %1, off"
                 :: "v"(lds), "v"(g) : "memory");
}
static __device__ __forceinline__ void wait_async0() {
    asm volatile("s_wait_asynccnt 0x0" ::: "memory");
}

// ---------------------------------------------------------------- mean over seq
__global__ __launch_bounds__(256) void mean_rows_kernel(
    const float* __restrict__ in, float* __restrict__ out) {
    int idx = blockIdx.x * 256 + threadIdx.x;          // over NSHOTS*D
    if (idx >= NSHOTS * D) return;
    int n = idx / D, j = idx % D;
    const float* p = in + (size_t)n * SEQ * D + j;
    float s = 0.f;
    #pragma unroll 4
    for (int t = 0; t < SEQ; ++t) s += p[(size_t)t * D];
    out[idx] = s * (1.0f / SEQ);
}

// -------------------------------------------- f32[K,N] -> bf16 transposed [N,K]
__global__ __launch_bounds__(256) void cvt_bf16_transpose_kernel(
    const float* __restrict__ w, unsigned short* __restrict__ wt) {
    __shared__ unsigned short tile[64][65];            // padded vs bank conflicts
    const int n0 = blockIdx.x * 64, k0 = blockIdx.y * 64;
    for (int i = threadIdx.x; i < 64 * 64; i += 256) {
        int r = i >> 6, c = i & 63;                    // k = k0+r, n = n0+c
        tile[c][r] = f32_to_bf16(w[(size_t)(k0 + r) * D + n0 + c]);
    }
    __syncthreads();
    for (int i = threadIdx.x; i < 64 * 64; i += 256) {
        int r = i >> 6, c = i & 63;                    // n = n0+r, k = k0+c
        wt[(size_t)(n0 + r) * D + k0 + c] = tile[r][c];
    }
}

// ---------------------------------------------------------------- layernorm -> bf16
__global__ __launch_bounds__(256) void layernorm_bf16_kernel(
    const float* __restrict__ x, const float* __restrict__ g,
    const float* __restrict__ b, unsigned short* __restrict__ out) {
    __shared__ float red[256];
    const int row = blockIdx.x, tid = threadIdx.x;
    const float* xr = x + (size_t)row * D;
    float s = 0.f, s2 = 0.f;
    for (int j = tid; j < D; j += 256) { float v = xr[j]; s += v; s2 += v * v; }
    red[tid] = s; __syncthreads();
    for (int o = 128; o > 0; o >>= 1) { if (tid < o) red[tid] += red[tid + o]; __syncthreads(); }
    float mean = red[0] * (1.0f / D); __syncthreads();
    red[tid] = s2; __syncthreads();
    for (int o = 128; o > 0; o >>= 1) { if (tid < o) red[tid] += red[tid + o]; __syncthreads(); }
    float var = red[0] * (1.0f / D) - mean * mean;
    float inv = rsqrtf(var + 1e-5f);
    for (int j = tid; j < D; j += 256) {
        float v = (xr[j] - mean) * inv * g[j] + b[j];
        out[(size_t)row * D + j] = f32_to_bf16(v);
    }
}

// ---------------------------------------------------------------- BF16 WMMA GEMM
// C[M,N] = A[M,K](bf16) @ Wt[N,K](bf16, pre-transposed) + bias[N], f32 accum.
#define BM 64
#define BN 64
#define BK 64

__device__ __forceinline__ void stage_tile_async(
    unsigned short* dA, unsigned short* dB,
    const unsigned short* __restrict__ A, const unsigned short* __restrict__ Wt,
    int m0b, int n0b, int M, int K, int k0, int tid) {
    const int r  = tid >> 2;                 // 0..63
    const int cg = (tid & 3) * 16;           // 0,16,32,48
    int agr = m0b + r; if (agr >= M) agr = M - 1;
    const unsigned short* ga = A + (size_t)agr * K + k0 + cg;
    unsigned la = lds_off(dA + r * BK + cg);
    async_b128(la,       ga);
    async_b128(la + 16u, ga + 8);
    const unsigned short* gb = Wt + (size_t)(n0b + r) * K + k0 + cg;
    unsigned lb = lds_off(dB + r * BK + cg);
    async_b128(lb,       gb);
    async_b128(lb + 16u, gb + 8);
}

__device__ __forceinline__ void compute_tile(
    const unsigned short* tA, const unsigned short* tB,
    int wr, int wc, int half, int lm, v8f& c0, v8f& c1) {
    #pragma unroll
    for (int kk = 0; kk < BK; kk += 32) {
        // A fragment (16x32): lane(half,lm): K = kk + 16*half + e
        const unsigned short* arow = &tA[(wr * 16 + lm) * BK + kk];
        v8bf alo = *(const v8bf*)&arow[16 * half];
        v8bf ahi = *(const v8bf*)&arow[16 * half + 8];
        v16bf a;
        #pragma unroll
        for (int i = 0; i < 8; ++i) { a[i] = alo[i]; a[i + 8] = ahi[i]; }
        // B fragments (32x16): lane holds col N=lm, K = kk + 16*half + e
        const unsigned short* b0r = &tB[(wc * 32 + lm) * BK + kk];
        const unsigned short* b1r = &tB[(wc * 32 + 16 + lm) * BK + kk];
        v8bf bl0 = *(const v8bf*)&b0r[16 * half];
        v8bf bh0 = *(const v8bf*)&b0r[16 * half + 8];
        v8bf bl1 = *(const v8bf*)&b1r[16 * half];
        v8bf bh1 = *(const v8bf*)&b1r[16 * half + 8];
        v16bf bv0, bv1;
        #pragma unroll
        for (int i = 0; i < 8; ++i) { bv0[i] = bl0[i]; bv0[i + 8] = bh0[i]; }
        #pragma unroll
        for (int i = 0; i < 8; ++i) { bv1[i] = bl1[i]; bv1[i + 8] = bh1[i]; }
        c0 = __builtin_amdgcn_wmma_f32_16x16x32_bf16(false, a, false, bv0,
                                                     (short)0, c0, false, false);
        c1 = __builtin_amdgcn_wmma_f32_16x16x32_bf16(false, a, false, bv1,
                                                     (short)0, c1, false, false);
    }
}

__global__ __launch_bounds__(256) void gemm_bf16_wmma_kernel(
    const unsigned short* __restrict__ A, const unsigned short* __restrict__ Wt,
    const float* __restrict__ bias, float* __restrict__ C,
    int M, int N, int K) {
    __shared__ __attribute__((aligned(16))) unsigned short sA[2][BM * BK];
    __shared__ __attribute__((aligned(16))) unsigned short sB[2][BN * BK];

    const int tid  = threadIdx.x;
    const int lane = tid & 31;
    const int wave = tid >> 5;       // 0..7
    const int wr   = wave >> 1;      // 0..3 : 16-row slab
    const int wc   = wave & 1;       // 0..1 : 32-col slab
    const int m0b  = blockIdx.y * BM;
    const int n0b  = blockIdx.x * BN;
    const int half = lane >> 4;      // 0/1
    const int lm   = lane & 15;

    v8f c0 = {}; v8f c1 = {};

    int buf = 0;
    stage_tile_async(sA[0], sB[0], A, Wt, m0b, n0b, M, K, 0, tid);
    wait_async0();
    __syncthreads();

    for (int k0 = 0; k0 < K; k0 += BK) {
        if (k0 + BK < K)
            stage_tile_async(sA[buf ^ 1], sB[buf ^ 1], A, Wt, m0b, n0b, M, K,
                             k0 + BK, tid);
        compute_tile(sA[buf], sB[buf], wr, wc, half, lm, c0, c1);
        wait_async0();           // next tile's LDS writes complete (this wave)
        __syncthreads();         // visible to all waves; safe to reuse old buf
        buf ^= 1;
    }

    // C/D layout: VGPR r, lane -> M = r + 8*half, N = lm
    #pragma unroll
    for (int r = 0; r < 8; ++r) {
        int gm = m0b + wr * 16 + r + 8 * half;
        if (gm < M) {
            int n0 = n0b + wc * 32 + lm;
            C[(size_t)gm * N + n0]      = c0[r] + bias[n0];
            C[(size_t)gm * N + n0 + 16] = c1[r] + bias[n0 + 16];
        }
    }
}

// ------------------------------------------- LN2 + GELU + [D,2] proj + gumbel
__global__ __launch_bounds__(256) void ln_gelu_proj_gumbel_kernel(
    const float* __restrict__ h, const float* __restrict__ g,
    const float* __restrict__ b, const float* __restrict__ w2,
    const float* __restrict__ b2, const float* __restrict__ u,
    float* __restrict__ mask) {
    __shared__ float red[256];
    const int row = blockIdx.x, tid = threadIdx.x;
    const float* hr = h + (size_t)row * D;
    float s = 0.f, s2 = 0.f;
    for (int j = tid; j < D; j += 256) { float v = hr[j]; s += v; s2 += v * v; }
    red[tid] = s; __syncthreads();
    for (int o = 128; o > 0; o >>= 1) { if (tid < o) red[tid] += red[tid + o]; __syncthreads(); }
    float mean = red[0] * (1.0f / D); __syncthreads();
    red[tid] = s2; __syncthreads();
    for (int o = 128; o > 0; o >>= 1) { if (tid < o) red[tid] += red[tid + o]; __syncthreads(); }
    float var = red[0] * (1.0f / D) - mean * mean;
    float inv = rsqrtf(var + 1e-5f);
    __syncthreads();

    float p0 = 0.f, p1 = 0.f;
    for (int j = tid; j < D; j += 256) {
        float v  = (hr[j] - mean) * inv * g[j] + b[j];
        float ge = 0.5f * v * (1.0f + erff(v * 0.70710678118654752f)); // exact gelu
        p0 += ge * w2[j * 2 + 0];
        p1 += ge * w2[j * 2 + 1];
    }
    red[tid] = p0; __syncthreads();
    for (int o = 128; o > 0; o >>= 1) { if (tid < o) red[tid] += red[tid + o]; __syncthreads(); }
    float l0 = 0.f;
    if (tid == 0) l0 = red[0] + b2[0];
    __syncthreads();
    red[tid] = p1; __syncthreads();
    for (int o = 128; o > 0; o >>= 1) { if (tid < o) red[tid] += red[tid + o]; __syncthreads(); }
    if (tid == 0) {
        float l1 = red[0] + b2[1];
        float u0 = u[row * 2 + 0], u1 = u[row * 2 + 1];
        float g0 = -logf(-logf(u0 + 1e-10f) + 1e-10f);
        float g1 = -logf(-logf(u1 + 1e-10f) + 1e-10f);
        // softmax((l+g)/tau)[1], tau=1
        mask[row] = 1.0f / (1.0f + expf((l0 + g0) - (l1 + g1)));
    }
}

// ---------------------------------------------------- exact top-k, ascending positions
// jax.lax.top_k semantics: descending value, ties -> lower index first.
__global__ __launch_bounds__(256) void topk_sorted_kernel(
    const float* __restrict__ mask, int n, int k,
    int* __restrict__ pos_out, float* __restrict__ posf_out) {
    __shared__ float m[TOKROWS];
    __shared__ unsigned char sel[TOKROWS];
    for (int i = threadIdx.x; i < n; i += 256) m[i] = mask[i];
    __syncthreads();
    for (int i = threadIdx.x; i < n; i += 256) {
        float mi = m[i];
        int rank = 0;
        for (int j = 0; j < n; ++j) {
            float mj = m[j];
            rank += (mj > mi) || (mj == mi && j < i);
        }
        sel[i] = (rank < k) ? 1 : 0;
    }
    __syncthreads();
    if (threadIdx.x == 0) {
        int c = 0;
        for (int i = 0; i < n; ++i)
            if (sel[i]) { pos_out[c] = i; posf_out[c] = (float)i; ++c; }
    }
}

// ---------------------------------------------------------------- row gather
__global__ __launch_bounds__(256) void gather_rows_kernel(
    const float* __restrict__ src, const int* __restrict__ pos,
    float* __restrict__ dst, int rpp) {
    int r = blockIdx.x;
    int p = pos[r / rpp];
    size_t srow = (size_t)p * rpp + (r % rpp);
    const float4* s = (const float4*)(src + srow * D);
    float4*       o = (float4*)(dst + (size_t)r * D);
    for (int j = threadIdx.x; j < D / 4; j += 256) o[j] = s[j];
}

// ================================================================ launcher
extern "C" void kernel_launch(void* const* d_in, const int* in_sizes, int n_in,
                              void* d_out, int out_size, void* d_ws, size_t ws_size,
                              hipStream_t stream) {
    const float* input_embeds = (const float*)d_in[0];
    const float* u_shot  = (const float*)d_in[1];
    const float* u_token = (const float*)d_in[2];
    const float* s_ln1g = (const float*)d_in[3];
    const float* s_ln1b = (const float*)d_in[4];
    const float* s_w1   = (const float*)d_in[5];
    const float* s_b1   = (const float*)d_in[6];
    const float* s_ln2g = (const float*)d_in[7];
    const float* s_ln2b = (const float*)d_in[8];
    const float* s_w2   = (const float*)d_in[9];
    const float* s_b2   = (const float*)d_in[10];
    const float* t_ln1g = (const float*)d_in[11];
    const float* t_ln1b = (const float*)d_in[12];
    const float* t_w1   = (const float*)d_in[13];
    const float* t_b1   = (const float*)d_in[14];
    const float* t_ln2g = (const float*)d_in[15];
    const float* t_ln2b = (const float*)d_in[16];
    const float* t_w2   = (const float*)d_in[17];
    const float* t_b2   = (const float*)d_in[18];

    float* out        = (float*)d_out;
    float* out_pruned = out;                                     // [4,512,4096]
    float* out_final  = out_pruned + (size_t)TGT_SHOT * SEQ * D; // [1024,4096]
    float* out_spos   = out_final + (size_t)TGT_TOK * D;         // [4]
    float* out_tpos   = out_spos + TGT_SHOT;                     // [1024]

    char* w = (char*)d_ws;
    unsigned short* w1s = (unsigned short*)w;  w += (size_t)D * D * 2;  // Wt shot [N,K]
    unsigned short* w1t = (unsigned short*)w;  w += (size_t)D * D * 2;  // Wt tok  [N,K]
    float* sent         = (float*)w;           w += (size_t)NSHOTS * D * 4;
    unsigned short* xln = (unsigned short*)w;  w += (size_t)TOKROWS * D * 2;
    float* h1           = (float*)w;           w += (size_t)TOKROWS * D * 4;
    float* maskbuf      = (float*)w;           w += (size_t)TOKROWS * 4;
    int* shotpos        = (int*)w;             w += 16;
    int* tokpos         = (int*)w;             w += (size_t)TGT_TOK * 4;

    // weight conversion + transpose: f32 [K,N] -> bf16 [N,K]
    cvt_bf16_transpose_kernel<<<dim3(D / 64, D / 64), 256, 0, stream>>>(s_w1, w1s);
    cvt_bf16_transpose_kernel<<<dim3(D / 64, D / 64), 256, 0, stream>>>(t_w1, w1t);

    // ---- shot stage ----
    mean_rows_kernel<<<(NSHOTS * D) / 256, 256, 0, stream>>>(input_embeds, sent);
    layernorm_bf16_kernel<<<NSHOTS, 256, 0, stream>>>(sent, s_ln1g, s_ln1b, xln);
    gemm_bf16_wmma_kernel<<<dim3(D / BN, (NSHOTS + BM - 1) / BM), 256, 0, stream>>>(
        xln, w1s, s_b1, h1, NSHOTS, D, D);
    ln_gelu_proj_gumbel_kernel<<<NSHOTS, 256, 0, stream>>>(
        h1, s_ln2g, s_ln2b, s_w2, s_b2, u_shot, maskbuf);
    topk_sorted_kernel<<<1, 256, 0, stream>>>(maskbuf, NSHOTS, TGT_SHOT, shotpos, out_spos);
    gather_rows_kernel<<<TGT_SHOT * SEQ, 256, 0, stream>>>(input_embeds, shotpos,
                                                           out_pruned, SEQ);

    // ---- token stage (reads pruned rows straight out of d_out) ----
    layernorm_bf16_kernel<<<TOKROWS, 256, 0, stream>>>(out_pruned, t_ln1g, t_ln1b, xln);
    gemm_bf16_wmma_kernel<<<dim3(D / BN, TOKROWS / BM), 256, 0, stream>>>(
        xln, w1t, t_b1, h1, TOKROWS, D, D);
    ln_gelu_proj_gumbel_kernel<<<TOKROWS, 256, 0, stream>>>(
        h1, t_ln2g, t_ln2b, t_w2, t_b2, u_token, maskbuf);
    topk_sorted_kernel<<<1, 256, 0, stream>>>(maskbuf, TOKROWS, TGT_TOK, tokpos, out_tpos);
    gather_rows_kernel<<<TGT_TOK, 256, 0, stream>>>(out_pruned, tokpos, out_final, 1);
}
